// QCProjectionCenterDiameterNN_55843164783020
// MI455X (gfx1250) — compile-verified, matrix operand order checked by reading, not run on hardware
//
#include <hip/hip_runtime.h>
#include <hip/hip_bf16.h>

// ---------------------------------------------------------------------------
// QCProjection — quadratic-constraint ray projection.
// Dominant op: rPr[s,c] = r_s^T P_c r_s  (S=4096, C=64, F=256, ~34 GFLOP)
// bf16-split (hi+lo) WMMA GEMM, f32 accumulation, async-to-LDS staging of P.
// ---------------------------------------------------------------------------

typedef __bf16 bf16;
typedef __attribute__((ext_vector_type(8)))  __bf16 bf16x8;
typedef __attribute__((ext_vector_type(16))) __bf16 bf16x16;
typedef __attribute__((ext_vector_type(8)))  float  f32x8;
typedef int i32x4 __attribute__((vector_size(16)));
typedef i32x4 __attribute__((address_space(3))) as3_i32x4;

#define F_DIM 256
#define C_DIM 64
#define SPB   128           // samples per block (8 waves x 16-row tiles)
#define CPB   2             // constraints per block
#define RSTR  264           // padded LDS row stride for R (bf16 elems)
#define BSTR  40            // padded LDS row stride for staged P slab (bf16)
#define BHALF (64 * BSTR)   // one hi or lo staging plane (2560 bf16 elems)
#define BBUF  (2 * BHALF)   // one double-buffer slot (hi+lo)
#define EPSV  1e-12f

// --- CDNA5 async copy global->LDS (ASYNCcnt tracked) -----------------------
#if defined(__has_builtin) && __has_builtin(__builtin_amdgcn_global_load_async_to_lds_b128)
#define ASYNC_B128(gp, lp)                                                   \
  __builtin_amdgcn_global_load_async_to_lds_b128(                            \
      (i32x4*)(void*)(gp), (as3_i32x4*)(lp), 0, 0)
#else
#define ASYNC_B128(gp, lp)                                                   \
  asm volatile("global_load_async_to_lds_b128 %0, %1, off"                   \
               :: "v"((unsigned)(unsigned long long)(lp)),                    \
                  "v"((unsigned long long)(gp)) : "memory")
#endif

#if defined(__has_builtin) && __has_builtin(__builtin_amdgcn_s_wait_asynccnt)
#define WAIT_ASYNC(n) __builtin_amdgcn_s_wait_asynccnt(n)
#else
#define WAIT_ASYNC(n) asm volatile("s_wait_asynccnt %0" :: "i"(n) : "memory")
#endif

static __device__ __forceinline__ bf16x16 concat8(bf16x8 a, bf16x8 b) {
  return __builtin_shufflevector(a, b, 0,1,2,3,4,5,6,7,8,9,10,11,12,13,14,15);
}

// ---------------------------------------------------------------------------
// Kernel 1: per-sample prep. ray = zs - point, length, norm ray, bf16 hi/lo
// split of norm ray, init max_length to +inf.  grid = S, block = 256 (= F).
// ---------------------------------------------------------------------------
__global__ __launch_bounds__(256)
void prep_sample_kernel(const float* __restrict__ zs, const float* __restrict__ point,
                        float* __restrict__ rn, bf16* __restrict__ Rhi, bf16* __restrict__ Rlo,
                        float* __restrict__ len, unsigned int* __restrict__ ml) {
  __shared__ float red[256];
  const int s = blockIdx.x;
  const int t = threadIdx.x;
  const float ray = zs[s * F_DIM + t] - point[t];
  red[t] = ray * ray;
  __syncthreads();
  for (int off = 128; off > 0; off >>= 1) {
    if (t < off) red[t] += red[t + off];
    __syncthreads();
  }
  const float L = fmaxf(sqrtf(red[0]), EPSV);
  const float r = ray / L;
  rn[s * F_DIM + t] = r;
  const bf16 h = (bf16)r;
  Rhi[s * F_DIM + t] = h;
  Rlo[s * F_DIM + t] = (bf16)(r - (float)h);
  if (t == 0) { len[s] = L; ml[s] = 0x7F800000u; /* +inf bits */ }
}

// ---------------------------------------------------------------------------
// Kernel 2: per-constraint prep.
//   v[c,k] = sum_f point[f]*P[c,k,f] + q[c,k]   (tPr+qr = v_c . r_s)
//   g[c]   = q_c.point + 0.5*tPt - b[c]
// ---------------------------------------------------------------------------
__global__ __launch_bounds__(256)
void prep_constraint_kernel(const float* __restrict__ P, const float* __restrict__ q,
                            const float* __restrict__ b, const float* __restrict__ point,
                            float* __restrict__ v, float* __restrict__ g) {
  __shared__ float red[256];
  const int c = blockIdx.x;
  const int k = threadIdx.x;
  const float* Prow = P + ((size_t)c * F_DIM + k) * F_DIM;
  float w = 0.f;
  for (int f = 0; f < F_DIM; ++f) w = fmaf(point[f], Prow[f], w);
  const float qv = q[c * F_DIM + k];
  v[c * F_DIM + k] = w + qv;
  red[k] = (qv + 0.5f * w) * point[k];
  __syncthreads();
  for (int off = 128; off > 0; off >>= 1) {
    if (k < off) red[k] += red[k + off];
    __syncthreads();
  }
  if (k == 0) g[c] = red[0] - b[c];
}

// ---------------------------------------------------------------------------
// Kernel 3: split P (f32) into bf16 hi + lo residual, same [c][k][f] layout.
// ---------------------------------------------------------------------------
__global__ __launch_bounds__(256)
void conv_P_kernel(const float* __restrict__ P, bf16* __restrict__ Phi,
                   bf16* __restrict__ Plo, int n) {
  const int i = blockIdx.x * 256 + threadIdx.x;
  if (i >= n) return;
  const float p = P[i];
  const bf16 h = (bf16)p;
  Phi[i] = h;
  Plo[i] = (bf16)(p - (float)h);
}

// ---------------------------------------------------------------------------
// Kernel 4: main WMMA quadratic-form kernel with async-to-LDS P staging.
// grid = (S/128, C/CPB), block = 256 threads (8 waves, 16-sample tile each).
// ---------------------------------------------------------------------------
__global__ __launch_bounds__(256)
void qform_wmma_kernel(const bf16* __restrict__ Rhi_g, const bf16* __restrict__ Rlo_g,
                       const bf16* __restrict__ Phi_g, const bf16* __restrict__ Plo_g,
                       const float* __restrict__ v_g, const float* __restrict__ g_g,
                       unsigned int* __restrict__ ml) {
  extern __shared__ char smem[];
  bf16*  sRhi = (bf16*)smem;                       // [SPB][RSTR]
  bf16*  sRlo = sRhi + SPB * RSTR;                 // [SPB][RSTR]
  float* sV   = (float*)(sRlo + SPB * RSTR);       // [F_DIM]
  bf16*  sB   = (bf16*)(sV + F_DIM);               // staging: 2 bufs x (hi+lo) x 64x40

  const int tid = threadIdx.x;
  const int s0  = blockIdx.x * SPB;
  const int c0  = blockIdx.y * CPB;

  // Cooperative load of the R tile (hi & lo) into padded LDS.
  for (int idx = tid; idx < SPB * (F_DIM / 8); idx += 256) {
    const int row = idx >> 5;            // F_DIM/8 = 32 chunks per row
    const int col = (idx & 31) * 8;
    *(bf16x8*)&sRhi[row * RSTR + col] = *(const bf16x8*)&Rhi_g[(size_t)(s0 + row) * F_DIM + col];
    *(bf16x8*)&sRlo[row * RSTR + col] = *(const bf16x8*)&Rlo_g[(size_t)(s0 + row) * F_DIM + col];
  }

  const int lane = tid & 31;
  const int wave = tid >> 5;
  const int half = lane >> 4;            // 0: lanes 0-15, 1: lanes 16-31
  const int l16  = lane & 15;
  const int m0   = wave * 16;            // this wave's 16 sample rows in tile
  const int rr   = tid >> 2;             // staging: row 0..63
  const int ch   = tid & 3;              // staging: 8-elem chunk 0..3

  for (int ci = 0; ci < CPB; ++ci) {
    const int c = c0 + ci;
    __syncthreads();                     // sV / staging reuse fence across c
    sV[tid] = v_g[c * F_DIM + tid];

    const bf16* Pc_hi = Phi_g + (size_t)c * F_DIM * F_DIM;
    const bf16* Pc_lo = Plo_g + (size_t)c * F_DIM * F_DIM;

    // issue one staging stage: 64 rows x 32 f of P_c (hi & lo) -> LDS buf
    auto issue = [&](int st) {
      const int ntg = st >> 3, kk = st & 7, buf = st & 1;
      const unsigned le = (unsigned)(buf * BBUF + rr * BSTR + ch * 8);
      const size_t   ge = (size_t)(ntg * 64 + rr) * F_DIM + kk * 32 + ch * 8;
      ASYNC_B128(Pc_hi + ge, sB + le);
      ASYNC_B128(Pc_lo + ge, sB + le + BHALF);
    };

    float epi[8];
#pragma unroll
    for (int j = 0; j < 8; ++j) epi[j] = 0.f;

    f32x8 acc[4];

    issue(0);                            // prologue
#pragma unroll 1
    for (int st = 0; st < 32; ++st) {    // 4 ntile-groups x 8 k-steps
      const int kk  = st & 7;
      const int buf = st & 1;
      if (kk == 0) {
#pragma unroll
        for (int nt = 0; nt < 4; ++nt)
#pragma unroll
          for (int j = 0; j < 8; ++j) acc[nt][j] = 0.f;
      }
      if (st + 1 < 32) { issue(st + 1); WAIT_ASYNC(2); }
      else             { WAIT_ASYNC(0); }
      __syncthreads();                   // stage st data visible to all waves

      // A fragments (ISA 16-bit layout: two 8-elem chunks per lane)
      const int f0   = kk * 32;
      const int fo0  = f0 + half * 8;
      const int fo1  = f0 + 16 + half * 8;
      const int arow = m0 + l16;
      const bf16x16 ahi = concat8(*(bf16x8*)&sRhi[arow * RSTR + fo0],
                                  *(bf16x8*)&sRhi[arow * RSTR + fo1]);
      const bf16x16 alo = concat8(*(bf16x8*)&sRlo[arow * RSTR + fo0],
                                  *(bf16x8*)&sRlo[arow * RSTR + fo1]);
#pragma unroll
      for (int nt = 0; nt < 4; ++nt) {
        const bf16* bb = sB + buf * BBUF + (nt * 16 + l16) * BSTR;
        const bf16x16 bhi = concat8(*(bf16x8*)&bb[half * 8],
                                    *(bf16x8*)&bb[16 + half * 8]);
        const bf16x16 blo = concat8(*(bf16x8*)&bb[BHALF + half * 8],
                                    *(bf16x8*)&bb[BHALF + 16 + half * 8]);
        // split-precision accumulate: hi*hi + hi*lo + lo*hi
        acc[nt] = __builtin_amdgcn_wmma_f32_16x16x32_bf16(
            false, ahi, false, bhi, (short)0, acc[nt], false, false);
        acc[nt] = __builtin_amdgcn_wmma_f32_16x16x32_bf16(
            false, ahi, false, blo, (short)0, acc[nt], false, false);
        acc[nt] = __builtin_amdgcn_wmma_f32_16x16x32_bf16(
            false, alo, false, bhi, (short)0, acc[nt], false, false);
      }
      __syncthreads();                   // all waves done reading buf[st&1]

      if (kk == 7) {
        // fused epilogue: rPr partial = sum_k Y[s,k]*r[s,k] over this n-slab
        const int ntg = st >> 3;
#pragma unroll
        for (int nt = 0; nt < 4; ++nt) {
          const int ncol = (ntg * 4 + nt) * 16 + l16;
#pragma unroll
          for (int j = 0; j < 8; ++j) {
            // D layout: VGPR j holds M = j + 8*half, N = lane%16
            const int row = m0 + half * 8 + j;
            const float rv = (float)sRhi[row * RSTR + ncol] + (float)sRlo[row * RSTR + ncol];
            epi[j] = fmaf(acc[nt][j], rv, epi[j]);
          }
        }
      }
    }

    // butterfly reduce over the 16 lanes of each half
#pragma unroll
    for (int j = 0; j < 8; ++j) {
      float t = epi[j];
      t += __shfl_xor(t, 1, 32);
      t += __shfl_xor(t, 2, 32);
      t += __shfl_xor(t, 4, 32);
      t += __shfl_xor(t, 8, 32);
      epi[j] = t;
    }

    // 16 lanes per wave each finish one sample
    const float gc = g_g[c];
    if (l16 < 8) {
      const int row   = m0 + half * 8 + l16;
      const float rPr = epi[l16];
      float dot = 0.f;                         // v_c . r_s
      for (int f = 0; f < F_DIM; f += 8) {
        const bf16x8 h = *(bf16x8*)&sRhi[row * RSTR + f];
        const bf16x8 l = *(bf16x8*)&sRlo[row * RSTR + f];
#pragma unroll
        for (int u = 0; u < 8; ++u)
          dot = fmaf(sV[f + u], (float)h[u] + (float)l[u], dot);
      }
      const float d4 = dot * dot - rPr * (2.0f * gc);
      const float aq = (-dot + sqrtf(d4)) / rPr;
      const float al = -gc / dot;
      float mu = (fabsf(rPr) <= EPSV) ? al : aq;
      const float mupos = (mu >= 0.f) ? mu : __builtin_inff();
      // all candidates >= 0 (or +inf): float-bit order == uint order
      atomicMin(&ml[s0 + row], __float_as_uint(mupos));
    }
  }
}

// ---------------------------------------------------------------------------
// Kernel 5: xs = point + min(length, max_length) * norm_ray
// ---------------------------------------------------------------------------
__global__ __launch_bounds__(256)
void finalize_kernel(const float* __restrict__ rn, const float* __restrict__ len,
                     const unsigned int* __restrict__ ml, const float* __restrict__ point,
                     float* __restrict__ xs, int n) {
  const int i = blockIdx.x * 256 + threadIdx.x;
  if (i >= n) return;
  const int s = i >> 8;
  const int f = i & (F_DIM - 1);
  const float L = fminf(len[s], __uint_as_float(ml[s]));
  xs[i] = point[f] + L * rn[i];
}

// ---------------------------------------------------------------------------
extern "C" void kernel_launch(void* const* d_in, const int* in_sizes, int n_in,
                              void* d_out, int out_size, void* d_ws, size_t ws_size,
                              hipStream_t stream) {
  const float* zs    = (const float*)d_in[0];   // (S, F)
  const float* P     = (const float*)d_in[1];   // (C, F, F)
  const float* q     = (const float*)d_in[2];   // (C, F)
  const float* b     = (const float*)d_in[3];   // (C,)
  const float* point = (const float*)d_in[4];   // (F,)
  float* xs = (float*)d_out;

  const int S = in_sizes[0] / F_DIM;            // 4096
  const int C = in_sizes[3];                    // 64

  // workspace layout (bytes, 16B aligned blocks)
  char* w = (char*)d_ws;
  size_t off = 0;
  float*        v_ws   = (float*)(w + off);  off += (size_t)C * F_DIM * 4;           // 64 KB
  float*        g_ws   = (float*)(w + off);  off += 1024;                            // C*4 padded
  float*        len_ws = (float*)(w + off);  off += (size_t)S * 4;                   // 16 KB
  unsigned int* ml_ws  = (unsigned int*)(w + off); off += (size_t)S * 4;             // 16 KB
  bf16*         Rhi_ws = (bf16*)(w + off);   off += (size_t)S * F_DIM * 2;           // 2 MB
  bf16*         Rlo_ws = (bf16*)(w + off);   off += (size_t)S * F_DIM * 2;           // 2 MB
  float*        rn_ws  = (float*)(w + off);  off += (size_t)S * F_DIM * 4;           // 4 MB
  bf16*         Phi_ws = (bf16*)(w + off);   off += (size_t)C * F_DIM * F_DIM * 2;   // 8.4 MB
  bf16*         Plo_ws = (bf16*)(w + off);   off += (size_t)C * F_DIM * F_DIM * 2;   // 8.4 MB
  (void)ws_size; (void)n_in; (void)out_size;

  // 1) per-sample prep
  prep_sample_kernel<<<S, 256, 0, stream>>>(zs, point, rn_ws, Rhi_ws, Rlo_ws, len_ws, ml_ws);
  // 2) per-constraint prep
  prep_constraint_kernel<<<C, 256, 0, stream>>>(P, q, b, point, v_ws, g_ws);
  // 3) P -> bf16 hi/lo
  const int nP = C * F_DIM * F_DIM;
  conv_P_kernel<<<(nP + 255) / 256, 256, 0, stream>>>(P, Phi_ws, Plo_ws, nP);
  // 4) main WMMA quadratic form + constraint intersection
  const int mainLds = 2 * SPB * RSTR * (int)sizeof(bf16)   // R hi/lo tiles
                    + F_DIM * (int)sizeof(float)           // sV
                    + 2 * BBUF * (int)sizeof(bf16);        // staging double buffer
  dim3 grid(S / SPB, C / CPB);
  qform_wmma_kernel<<<grid, 256, mainLds, stream>>>(Rhi_ws, Rlo_ws, Phi_ws, Plo_ws,
                                                    v_ws, g_ws, ml_ws);
  // 5) final projection
  const int nOut = S * F_DIM;
  finalize_kernel<<<(nOut + 255) / 256, 256, 0, stream>>>(rn_ws, len_ws, ml_ws, point, xs, nOut);
}
